// PAM_Module_33741263077582
// MI455X (gfx1250) — matrix operand order repaired and checked by reading, not compiled, not run on hardware
//
#include <hip/hip_runtime.h>
#include <hip/hip_bf16.h>

// ---------------- types ----------------
typedef __bf16 v16bf __attribute__((ext_vector_type(16)));
typedef __bf16 v8bf  __attribute__((ext_vector_type(8)));
typedef __bf16 v2bf  __attribute__((ext_vector_type(2)));
typedef float  v8f   __attribute__((ext_vector_type(8)));

static constexpr int Bb = 8;
static constexpr int Ll = 4096;
static constexpr int Cc = 256;
static constexpr int Dd = 32;   // C8
static constexpr float SHIFT = 8.0f;   // fixed softmax shift (ratio-invariant)

#if __has_builtin(__builtin_amdgcn_cvt_pk_bf16_f32)
#define HAVE_PK_BF16 1
#else
#define HAVE_PK_BF16 0
#endif

__device__ __forceinline__ unsigned short f2bf(float f) {
    unsigned u = __float_as_uint(f);
    unsigned r = u + 0x7FFFu + ((u >> 16) & 1u);   // round-to-nearest-even
    return (unsigned short)(r >> 16);
}

// pack two floats into two bf16 (low = a, high = b)
__device__ __forceinline__ unsigned pack2bf(float a, float b) {
#if HAVE_PK_BF16
    union { v2bf v; unsigned u; } cv;
    cv.v = __builtin_amdgcn_cvt_pk_bf16_f32(a, b);
    return cv.u;
#else
    return (unsigned)f2bf(a) | ((unsigned)f2bf(b) << 16);
#endif
}

__device__ __forceinline__ v16bf cat8(v8bf lo, v8bf hi) {
    return __builtin_shufflevector(lo, hi, 0,1,2,3,4,5,6,7,8,9,10,11,12,13,14,15);
}

// ---------------- stage 0: fp32 -> bf16 conversion (2 elems/thread) ----------------
__global__ __launch_bounds__(256) void cvt_bf16_kernel(
    const float* __restrict__ src, unsigned short* __restrict__ dst)
{
    int tid = blockIdx.x * 256 + threadIdx.x;
    float2 v = ((const float2*)src)[tid];
    ((unsigned*)dst)[tid] = pack2bf(v.x, v.y);
}

// ---------------- stage 1a: Q/K projections via WMMA -> bf16 [B*L, 32] ----------
// Wave = one 16-point l-tile; 4 M-tiles (Wq rows 0-15,16-31; Wk rows 0-15,16-31),
// K = 256 in 8 chunks of 32.
__global__ __launch_bounds__(128) void proj_qk_kernel(
    const unsigned short* __restrict__ xbf,
    const unsigned short* __restrict__ Wqb, const unsigned short* __restrict__ Wkb,
    const float* __restrict__ bq, const float* __restrict__ bk,
    unsigned short* __restrict__ Qb, unsigned short* __restrict__ Kb)
{
    const int lane = threadIdx.x & 31;
    const int w    = threadIdx.x >> 5;
    const int half = lane >> 4;
    const int n    = lane & 15;

    const int p = (blockIdx.x * 4 + w) * 16 + n;         // global point for this lane
    const unsigned short* xr = xbf + (size_t)p * Cc;

    v8f acc[4];
    #pragma unroll
    for (int mt = 0; mt < 4; ++mt) acc[mt] = v8f{};

    for (int cc = 0; cc < Cc; cc += 32) {
        // B operand: x^T chunk, lane col = point n, K = cc + 16h + e (contiguous)
        const unsigned short* xp = xr + cc + 16 * half;
        v16bf Bx = cat8(*(const v8bf*)xp, *(const v8bf*)(xp + 8));
        #pragma unroll
        for (int mt = 0; mt < 4; ++mt) {
            const unsigned short* wr = (mt < 2 ? Wqb + (size_t)(mt * 16 + n) * Cc
                                               : Wkb + (size_t)((mt - 2) * 16 + n) * Cc);
            v16bf Aw = cat8(*(const v8bf*)(wr + cc + 16 * half),
                            *(const v8bf*)(wr + cc + 16 * half + 8));
            acc[mt] = __builtin_amdgcn_wmma_f32_16x16x32_bf16(false, Aw, false, Bx, (short)0, acc[mt], false, false);
        }
    }

    // bias + store: D layout row = dim d, col = point; consecutive r -> consecutive d
    #pragma unroll
    for (int mt = 0; mt < 4; ++mt) {
        unsigned short* base = (mt < 2) ? Qb : Kb;
        const float* bias = (mt < 2) ? bq : bk;
        const int dbase = (mt & 1) * 16;
        #pragma unroll
        for (int r = 0; r < 8; r += 2) {
            int d = dbase + r + 8 * half;
            float v0 = acc[mt][r]     + bias[d];
            float v1 = acc[mt][r + 1] + bias[d + 1];
            *(unsigned*)&base[(size_t)p * Dd + d] = pack2bf(v0, v1);
        }
    }
}

// ---------------- stage 1b: V projection via WMMA -> Vt[B][C][L] bf16 -----------
// Keys interleaved within each 32-chunk: offset o stored at k = (o<16)?2o:2(o-16)+1,
// matching the attention kernel's packed-b32 P spill.
// Wave = one 16-point l-tile x all 16 channel tiles; K = 256 in 8 chunks.
__global__ __launch_bounds__(128) void proj_v_kernel(
    const unsigned short* __restrict__ xbf, const unsigned short* __restrict__ Wvb,
    const float* __restrict__ bv, unsigned short* __restrict__ Vt)
{
    const int lane = threadIdx.x & 31;
    const int w    = threadIdx.x >> 5;
    const int half = lane >> 4;
    const int n    = lane & 15;

    const int p = (blockIdx.x * 4 + w) * 16 + n;         // global point
    const int b    = p >> 12;
    const int lloc = p & (Ll - 1);
    const unsigned short* xr = xbf + (size_t)p * Cc;

    v8f acc[16];
    #pragma unroll
    for (int ct = 0; ct < 16; ++ct) acc[ct] = v8f{};

    for (int cc = 0; cc < Cc; cc += 32) {
        const unsigned short* xp = xr + cc + 16 * half;
        v16bf Bx = cat8(*(const v8bf*)xp, *(const v8bf*)(xp + 8));
        #pragma unroll
        for (int ct = 0; ct < 16; ++ct) {
            const unsigned short* wr = Wvb + (size_t)(ct * 16 + n) * Cc;
            v16bf Aw = cat8(*(const v8bf*)(wr + cc + 16 * half),
                            *(const v8bf*)(wr + cc + 16 * half + 8));
            acc[ct] = __builtin_amdgcn_wmma_f32_16x16x32_bf16(false, Aw, false, Bx, (short)0, acc[ct], false, false);
        }
    }

    // bias + transposed, key-interleaved store
    const int slot = 2 * n + ((lloc >> 4) & 1);          // position within 32-key chunk
    const size_t lchunk = (size_t)(lloc & ~31);
    #pragma unroll
    for (int ct = 0; ct < 16; ++ct) {
        #pragma unroll
        for (int r = 0; r < 8; ++r) {
            int c = ct * 16 + r + 8 * half;
            float val = acc[ct][r] + bv[c];
            Vt[((size_t)(b * Cc + c)) * Ll + lchunk + slot] = f2bf(val);
        }
    }
}

// ---------------- stage 2: attention, O[B][L][C] fp32 ----------------
// Block = 128 threads (4 waves), 64 query rows per block.
// Wave w: computes S/exp for M-tile w (16 rows), owns output channels
// [w*64, w*64+64) for ALL 4 M-tiles. P shared via double-buffered LDS,
// one barrier per 32-key chunk. Softmax denominator via a P x ones WMMA.
__global__ __launch_bounds__(128) void attn_kernel(
    const unsigned short* __restrict__ Qb, const unsigned short* __restrict__ Kb,
    const unsigned short* __restrict__ Vt, float* __restrict__ O)
{
    __shared__ __align__(16) unsigned short P[2][4][16][40];  // [buf][mtile][row][key(+pad)]
    __shared__ float Lsum[64];

    const int lane = threadIdx.x & 31;
    const int w    = threadIdx.x >> 5;
    const int half = lane >> 4;
    const int n    = lane & 15;

    const int bidx = blockIdx.x >> 6;           // 64 row-blocks per batch
    const int qblk = (blockIdx.x & 63) * 64;

    const unsigned short* qrow = Qb + ((size_t)(bidx * Ll + qblk + w * 16 + n)) * Dd;
    v16bf Qa = cat8(*(const v8bf*)(qrow + 16 * half),
                    *(const v8bf*)(qrow + 16 * half + 8));

    union { unsigned short u[16]; v16bf v; } ones_u;
    #pragma unroll
    for (int e = 0; e < 16; ++e) ones_u.u[e] = 0x3F80;
    const v16bf onesB = ones_u.v;

    v8f acc[4][4];
    #pragma unroll
    for (int mt = 0; mt < 4; ++mt)
        #pragma unroll
        for (int t = 0; t < 4; ++t)
            acc[mt][t] = v8f{};
    v8f lacc = {};

    const unsigned short* kbase = Kb + (size_t)bidx * Ll * Dd;
    const unsigned short* vbase = Vt + (size_t)bidx * Cc * Ll;
    const int cb = w * 64;

    for (int j = 0; j < Ll; j += 32) {
        const int buf = (j >> 5) & 1;
        if (j + 32 < Ll)
            __builtin_prefetch(kbase + (size_t)(j + 32 + n) * Dd, 0, 1);

        const unsigned short* k0 = kbase + (size_t)(j + n) * Dd + 16 * half;
        const unsigned short* k1 = kbase + (size_t)(j + 16 + n) * Dd + 16 * half;
        v16bf Kb0 = cat8(*(const v8bf*)k0, *(const v8bf*)(k0 + 8));
        v16bf Kb1 = cat8(*(const v8bf*)k1, *(const v8bf*)(k1 + 8));

        v8f z = {};
        v8f S0 = __builtin_amdgcn_wmma_f32_16x16x32_bf16(false, Qa, false, Kb0, (short)0, z, false, false);
        v8f S1 = __builtin_amdgcn_wmma_f32_16x16x32_bf16(false, Qa, false, Kb1, (short)0, z, false, false);

        // exp (fixed shift) + packed b32 spill: S0 col n -> slot 2n, S1 -> 2n+1
        #pragma unroll
        for (int r = 0; r < 8; ++r) {
            unsigned pk = pack2bf(__expf(S0[r] - SHIFT), __expf(S1[r] - SHIFT));
            *(unsigned*)&P[buf][w][r + 8 * half][2 * n] = pk;
        }
        __syncthreads();

        v16bf Pa[4];
        #pragma unroll
        for (int mt = 0; mt < 4; ++mt) {
            const unsigned short* prow = &P[buf][mt][n][0];
            Pa[mt] = cat8(*(const v8bf*)(prow + 16 * half),
                          *(const v8bf*)(prow + 16 * half + 8));
        }

        lacc = __builtin_amdgcn_wmma_f32_16x16x32_bf16(false, Pa[w], false, onesB, (short)0, lacc, false, false);

        #pragma unroll
        for (int t = 0; t < 4; ++t) {
            const unsigned short* vp = vbase + (size_t)(cb + t * 16 + n) * Ll + j + 16 * half;
            v16bf Vb = cat8(*(const v8bf*)vp, *(const v8bf*)(vp + 8));
            #pragma unroll
            for (int mt = 0; mt < 4; ++mt)
                acc[mt][t] = __builtin_amdgcn_wmma_f32_16x16x32_bf16(false, Pa[mt], false, Vb, (short)0, acc[mt][t], false, false);
        }
    }

    if (n == 0) {
        #pragma unroll
        for (int r = 0; r < 8; ++r)
            Lsum[w * 16 + r + 8 * half] = lacc[r];
    }
    __syncthreads();

    #pragma unroll
    for (int mt = 0; mt < 4; ++mt) {
        #pragma unroll
        for (int r = 0; r < 8; ++r) {
            float inv = 1.0f / Lsum[mt * 16 + r + 8 * half];
            int row = qblk + mt * 16 + r + 8 * half;
            float* orow = O + ((size_t)bidx * Ll + row) * Cc + cb + n;
            #pragma unroll
            for (int t = 0; t < 4; ++t)
                orow[t * 16] = acc[mt][t][r] * inv;
        }
    }
}

// ---------------- stage 3: tiled-transpose combine ----------------
// y.flat[f] = gamma * O[f%L][f/L] + x.flat[f]  (per batch), i.e. y reads the
// [C][L]-transposed view of O. 32x32 LDS transpose: coalesced on both sides.
__global__ __launch_bounds__(256) void combine_kernel(
    const float* __restrict__ O, const float* __restrict__ x,
    const float* __restrict__ gamma, float* __restrict__ y)
{
    __shared__ float tile[32][33];
    const int tx = threadIdx.x & 31;
    const int ty = threadIdx.x >> 5;            // 0..7

    const int bpt = blockIdx.x;                 // b * 128*8 + lt * 8 + ct
    const int b  = bpt >> 10;
    const int lt = (bpt >> 3) & 127;
    const int ct = bpt & 7;
    const int l0 = lt * 32;
    const int c0 = ct * 32;
    const size_t obase = (size_t)b * Ll * Cc;
    const float g = gamma[0];

    #pragma unroll
    for (int s = 0; s < 4; ++s) {
        int ii = ty + 8 * s;
        tile[ii][tx] = O[obase + (size_t)(l0 + ii) * Cc + c0 + tx];
    }
    __syncthreads();
    #pragma unroll
    for (int s = 0; s < 4; ++s) {
        int jj = ty + 8 * s;
        size_t f = (size_t)(c0 + jj) * Ll + l0 + tx;   // output flat index in batch
        y[obase + f] = g * tile[tx][jj] + x[obase + f];
    }
}

// ---------------- host launcher ----------------
extern "C" void kernel_launch(void* const* d_in, const int* in_sizes, int n_in,
                              void* d_out, int out_size, void* d_ws, size_t ws_size,
                              hipStream_t stream)
{
    const float* feats = (const float*)d_in[0];
    const float* Wq    = (const float*)d_in[1];
    const float* bq    = (const float*)d_in[2];
    const float* Wk    = (const float*)d_in[3];
    const float* bk    = (const float*)d_in[4];
    const float* Wv    = (const float*)d_in[5];
    const float* bv    = (const float*)d_in[6];
    const float* gamma = (const float*)d_in[7];
    float* out = (float*)d_out;

    // workspace layout (all overwritten before read; ~68 MiB total)
    unsigned short* xbf  = (unsigned short*)d_ws;                 // 16 MiB
    unsigned short* Wqb  = xbf  + (size_t)Bb * Ll * Cc;           // 16 KiB
    unsigned short* Wkb  = Wqb  + (size_t)Dd * Cc;                // 16 KiB
    unsigned short* Wvb  = Wkb  + (size_t)Dd * Cc;                // 128 KiB
    unsigned short* Qbuf = Wvb  + (size_t)Cc * Cc;                //  2 MiB
    unsigned short* Kbuf = Qbuf + (size_t)Bb * Ll * Dd;           //  2 MiB
    unsigned short* Vt   = Kbuf + (size_t)Bb * Ll * Dd;           // 16 MiB
    float*          Ob   = (float*)(Vt + (size_t)Bb * Cc * Ll);   // 32 MiB

    // stage 0: bf16 conversions
    hipLaunchKernelGGL(cvt_bf16_kernel, dim3((Bb * Ll * Cc) / 512), dim3(256), 0, stream, feats, xbf);
    hipLaunchKernelGGL(cvt_bf16_kernel, dim3((Dd * Cc) / 512),      dim3(256), 0, stream, Wq, Wqb);
    hipLaunchKernelGGL(cvt_bf16_kernel, dim3((Dd * Cc) / 512),      dim3(256), 0, stream, Wk, Wkb);
    hipLaunchKernelGGL(cvt_bf16_kernel, dim3((Cc * Cc) / 512),      dim3(256), 0, stream, Wv, Wvb);

    // stage 1: WMMA projections
    hipLaunchKernelGGL(proj_qk_kernel, dim3((Bb * Ll) / 64), dim3(128), 0, stream,
                       xbf, Wqb, Wkb, bq, bk, Qbuf, Kbuf);
    hipLaunchKernelGGL(proj_v_kernel, dim3((Bb * Ll) / 64), dim3(128), 0, stream,
                       xbf, Wvb, bv, Vt);

    // stage 2: attention
    hipLaunchKernelGGL(attn_kernel, dim3(Bb * (Ll / 64)), dim3(128), 0, stream,
                       Qbuf, Kbuf, Vt, Ob);

    // stage 3: combine (tiled transpose)
    hipLaunchKernelGGL(combine_kernel, dim3(Bb * (Ll / 32) * (Cc / 32)), dim3(256), 0, stream,
                       Ob, feats, gamma, out);
}
